// RetardedNeighborDiscriminator_49898930045647
// MI455X (gfx1250) — compile-verified, matrix operand order checked by reading, not run on hardware
//
#include <hip/hip_runtime.h>
#include <hip/hip_bf16.h>

// ---------------------------------------------------------------------------
// RetardedNeighborDiscriminator for MI455X (gfx1250, wave32, WMMA)
// out[m] = max_n ( -K * sqrt(| ||X_n||^2 - 2 X_n.Xt_m + ||Xt_m||^2 |) + w[n] )
//
// Fast path: one-shot fp32->bf16 conversion pass into workspace, then a
// double-buffered bf16 WMMA GEMM whose next-tile fill uses gfx1250
// GLOBAL_LOAD_ASYNC_TO_LDS_B128 (ASYNCcnt-tracked, zero VGPR staging -> no
// spills), with a fused distance/max epilogue.
// ---------------------------------------------------------------------------

typedef __bf16 bf16x8  __attribute__((ext_vector_type(8)));
typedef __bf16 bf16x16 __attribute__((ext_vector_type(16)));
typedef float  f32x8   __attribute__((ext_vector_type(8)));
typedef int    i32x4   __attribute__((ext_vector_type(4)));

#define BM 128          // rows of X per block
#define BN 128          // rows of X_tilde (cols of C) per block
#define BK 64           // K elements staged per LDS buffer
#define LDW 36          // dwords per LDS tile row (72 bf16 = 64 + 8 pad) -> conflict-free
#define KSLOPE 10.0f

#if defined(__has_builtin)
#  if __has_builtin(__builtin_amdgcn_global_load_async_to_lds_b128)
#    define HAVE_ASYNC_LDS 1
#  endif
#endif
#ifndef HAVE_ASYNC_LDS
#  define HAVE_ASYNC_LDS 0
#endif

#define AS_GLOBAL __attribute__((address_space(1)))
#define AS_SHARED __attribute__((address_space(3)))

union FragU { bf16x16 v; bf16x8 h[2]; };

__device__ __forceinline__ unsigned pack_bf16x2(float a, float b) {
    // round-to-nearest-even fp32 -> bf16, packed pair
    unsigned ua = __float_as_uint(a), ub = __float_as_uint(b);
    ua += 0x7FFFu + ((ua >> 16) & 1u);
    ub += 0x7FFFu + ((ub >> 16) & 1u);
    return (ua >> 16) | (ub & 0xFFFF0000u);
}

// order-preserving float<->uint encoding (monotonic under unsigned max)
__device__ __forceinline__ unsigned fenc(float f) {
    unsigned u = __float_as_uint(f);
    return (u & 0x80000000u) ? ~u : (u | 0x80000000u);
}
__device__ __forceinline__ float fdec(unsigned u) {
    unsigned v = (u & 0x80000000u) ? (u ^ 0x80000000u) : ~u;
    return __uint_as_float(v);
}

// ---------------------------------------------------------------------------
// Kernel 0: init per-column running max (encoding 0 == below every real float)
// ---------------------------------------------------------------------------
__global__ void init_colmax(unsigned* __restrict__ colmax, int M) {
    int i = blockIdx.x * 256 + threadIdx.x;
    if (i < M) colmax[i] = 0u;
}

// ---------------------------------------------------------------------------
// Kernel 1: row sum-of-squares in fp32 for both X (rows [0,N)) and Xt ([N,N+M))
// ---------------------------------------------------------------------------
__global__ void row_sumsq(const float* __restrict__ X, const float* __restrict__ Xt,
                          float* __restrict__ xsq, float* __restrict__ ysq,
                          int N, int M, int D) {
    __shared__ float red[8];
    int row = blockIdx.x;
    const float* src;
    float* dst;
    if (row < N) { src = X  + (size_t)row * D;        dst = xsq + row; }
    else         { src = Xt + (size_t)(row - N) * D;  dst = ysq + (row - N); }

    float s = 0.f;
    for (int i = threadIdx.x; i < D; i += 256) {
        float v = src[i];
        s = fmaf(v, v, s);
    }
    #pragma unroll
    for (int o = 16; o > 0; o >>= 1) s += __shfl_down(s, o, 32);
    int lane = threadIdx.x & 31, wave = threadIdx.x >> 5;
    if (lane == 0) red[wave] = s;
    __syncthreads();
    if (threadIdx.x == 0) {
        float t = 0.f;
        #pragma unroll
        for (int i = 0; i < 8; ++i) t += red[i];
        *dst = t;
    }
}

// ---------------------------------------------------------------------------
// Kernel 2: one-shot fp32 -> packed-bf16 conversion (8 elems / thread)
// ---------------------------------------------------------------------------
__global__ void convert_bf16(const float* __restrict__ src, unsigned* __restrict__ dst,
                             long long nElem) {
    long long i = ((long long)blockIdx.x * 256 + threadIdx.x) * 8;
    if (i >= nElem) return;
    const float4 v0 = *(const float4*)(src + i);
    const float4 v1 = *(const float4*)(src + i + 4);
    uint4 p;
    p.x = pack_bf16x2(v0.x, v0.y);
    p.y = pack_bf16x2(v0.z, v0.w);
    p.z = pack_bf16x2(v1.x, v1.y);
    p.w = pack_bf16x2(v1.z, v1.w);
    *(uint4*)(dst + (i >> 1)) = p;
}

// ---------------------------------------------------------------------------
// Fragment gather + 8x WMMA for one 32-wide k-step (shared by both GEMMs).
// A layout (16-bit, 16x32): lanes 0-15 rows M=0-15 own K=kb..kb+7,kb+16..kb+23;
// lanes 16-31 own K=kb+8..kb+15,kb+24..kb+31. B mirrors with N over lanes.
// C layout: VGPR r: lanes0-15 -> (M=r, N=lane); lanes16-31 -> (M=8+r, N=lane-16)
// ---------------------------------------------------------------------------
__device__ __forceinline__ void wmma_step(const unsigned* __restrict__ Asb,
                                          const unsigned* __restrict__ Bsb,
                                          int wrow, int wcol, int rlo, int kd,
                                          f32x8 acc[4][2]) {
    bf16x16 afrag[4];
    #pragma unroll
    for (int mt = 0; mt < 4; ++mt) {
        const unsigned* base = &Asb[(wrow * 64 + mt * 16 + rlo) * LDW + kd];
        FragU f;
        f.h[0] = *(const bf16x8*)(base);      // K = kb .. kb+7
        f.h[1] = *(const bf16x8*)(base + 8);  // K = kb+16 .. kb+23
        afrag[mt] = f.v;
    }
    bf16x16 bfrag[2];
    #pragma unroll
    for (int nt = 0; nt < 2; ++nt) {
        const unsigned* base = &Bsb[(wcol * 32 + nt * 16 + rlo) * LDW + kd];
        FragU f;
        f.h[0] = *(const bf16x8*)(base);
        f.h[1] = *(const bf16x8*)(base + 8);
        bfrag[nt] = f.v;
    }
    #pragma unroll
    for (int mt = 0; mt < 4; ++mt)
        #pragma unroll
        for (int nt = 0; nt < 2; ++nt)
            acc[mt][nt] = __builtin_amdgcn_wmma_f32_16x16x32_bf16(
                false, afrag[mt], false, bfrag[nt],
                (short)0, acc[mt][nt], false, false);
}

// Fused epilogue: d2 -> -K*sqrt(|d2|) + w[row], column max -> LDS -> global
__device__ __forceinline__ void dist_max_epilogue(f32x8 acc[4][2],
                                                  const float* __restrict__ w,
                                                  const float* __restrict__ xsq,
                                                  const float* __restrict__ ysq,
                                                  unsigned* __restrict__ colmax,
                                                  unsigned* __restrict__ cmax,
                                                  int mWave, int nWave, int nBase,
                                                  int lane, int wcol, int tid) {
    const int rlo   = lane & 15;
    const int mhalf = (lane & 16) ? 8 : 0;
    #pragma unroll
    for (int nt = 0; nt < 2; ++nt) {
        const int gcol = nWave + nt * 16 + rlo;
        const float yq = ysq[gcol];
        float best = -__builtin_inff();
        #pragma unroll
        for (int mt = 0; mt < 4; ++mt) {
            const int mrow0 = mWave + mt * 16 + mhalf;
            #pragma unroll
            for (int r = 0; r < 8; ++r) {
                const int grow = mrow0 + r;
                const float ip  = acc[mt][nt][r];
                const float d2  = xsq[grow] - 2.0f * ip + yq;
                const float act = fmaf(-KSLOPE, __builtin_sqrtf(__builtin_fabsf(d2)), w[grow]);
                best = fmaxf(best, act);
            }
        }
        best = fmaxf(best, __shfl_xor(best, 16, 32));  // fold half-wave row groups
        atomicMax(&cmax[wcol * 32 + nt * 16 + rlo], fenc(best));
    }
    __syncthreads();
    for (int i = tid; i < BN; i += 256)
        atomicMax(&colmax[nBase + i], cmax[i]);
}

// ---------------------------------------------------------------------------
// Kernel 3 (fast path): bf16-input GEMM, LDS double-buffered. Next-tile fill
// uses GLOBAL_LOAD_ASYNC_TO_LDS_B128 (no VGPR staging, overlaps the WMMAs).
// grid = (M/BN fast, N/BM slow): X slab hot in L2 across the M sweep,
// X_tilde (25 MB bf16) fully L2-resident -> HBM reads X essentially once.
// ---------------------------------------------------------------------------
__global__ __launch_bounds__(256)
void gemm_dist_max_bf16(const unsigned* __restrict__ Xb, const unsigned* __restrict__ Xtb,
                        const float* __restrict__ w, const float* __restrict__ xsq,
                        const float* __restrict__ ysq, unsigned* __restrict__ colmax,
                        int D) {
    __shared__ unsigned As[2][BM * LDW];
    __shared__ unsigned Bs[2][BN * LDW];
    __shared__ unsigned cmax[BN];

    const int tid  = threadIdx.x;
    const int lane = tid & 31;
    const int wave = tid >> 5;
    const int wrow = wave >> 2;           // 0..1 : 64-row band
    const int wcol = wave & 3;            // 0..3 : 32-col band

    const int mBase = blockIdx.y * BM;
    const int nBase = blockIdx.x * BN;
    const int mWave = mBase + wrow * 64;
    const int nWave = nBase + wcol * 32;
    const int Dw    = D >> 1;             // dwords per bf16 row

    for (int i = tid; i < BN; i += 256) cmax[i] = 0u;

    f32x8 acc[4][2];
    #pragma unroll
    for (int mt = 0; mt < 4; ++mt)
        #pragma unroll
        for (int nt = 0; nt < 2; ++nt)
            acc[mt][nt] = (f32x8)0.f;

    const int rlo   = lane & 15;
    const int khalf = (lane & 16) ? 8 : 0;

    // Per tile: 128 rows x 32 dwords = 1024 uint4 slots -> 4 per thread per matrix.
    auto copyTile = [&](int k0, int buf) {
        const int kd = k0 >> 1;
        #pragma unroll
        for (int it = 0; it < 4; ++it) {
            const int slot = it * 256 + tid;
            const int row  = slot >> 3;      // 8 uint4 per row
            const int c4   = slot & 7;
            const unsigned* ga = Xb  + (size_t)(mBase + row) * Dw + kd + c4 * 4;
            const unsigned* gb = Xtb + (size_t)(nBase + row) * Dw + kd + c4 * 4;
            unsigned* la = &As[buf][row * LDW + c4 * 4];
            unsigned* lb = &Bs[buf][row * LDW + c4 * 4];
#if HAVE_ASYNC_LDS
            __builtin_amdgcn_global_load_async_to_lds_b128(
                (AS_GLOBAL i32x4*)ga, (AS_SHARED i32x4*)la, 0, 0);
            __builtin_amdgcn_global_load_async_to_lds_b128(
                (AS_GLOBAL i32x4*)gb, (AS_SHARED i32x4*)lb, 0, 0);
#else
            *(uint4*)la = *(const uint4*)ga;   // short live range: no spills
            *(uint4*)lb = *(const uint4*)gb;
#endif
        }
    };
    auto waitCopy = []() {
#if HAVE_ASYNC_LDS
        asm volatile("s_wait_asynccnt 0x0" ::: "memory");
#endif
    };

    copyTile(0, 0);
    waitCopy();
    __syncthreads();

    const int S = D / BK;                 // 48 stages
    for (int s = 0; s < S; ++s) {
        const int cur  = s & 1;
        const bool more = (s + 1 < S);
        if (more) copyTile((s + 1) * BK, cur ^ 1);      // DMA overlaps the WMMAs
        #pragma unroll
        for (int kk = 0; kk < BK; kk += 32)
            wmma_step(As[cur], Bs[cur], wrow, wcol, rlo, (kk + khalf) >> 1, acc);
        if (more) waitCopy();
        __syncthreads();
    }

    dist_max_epilogue(acc, w, xsq, ysq, colmax, cmax, mWave, nWave, nBase, lane, wcol, tid);
}

// ---------------------------------------------------------------------------
// Kernel 3' (fallback, small workspace): convert-in-kernel fp32 GEMM.
// ---------------------------------------------------------------------------
__global__ __launch_bounds__(256)
void gemm_dist_max_f32(const float* __restrict__ X, const float* __restrict__ Xt,
                       const float* __restrict__ w, const float* __restrict__ xsq,
                       const float* __restrict__ ysq, unsigned* __restrict__ colmax,
                       int D) {
    __shared__ unsigned As[BM * LDW];
    __shared__ unsigned Bs[BN * LDW];
    __shared__ unsigned cmax[BN];

    const int tid  = threadIdx.x;
    const int lane = tid & 31;
    const int wave = tid >> 5;
    const int wrow = wave >> 2;
    const int wcol = wave & 3;

    const int mBase = blockIdx.y * BM;
    const int nBase = blockIdx.x * BN;
    const int mWave = mBase + wrow * 64;
    const int nWave = nBase + wcol * 32;

    for (int i = tid; i < BN; i += 256) cmax[i] = 0u;

    f32x8 acc[4][2];
    #pragma unroll
    for (int mt = 0; mt < 4; ++mt)
        #pragma unroll
        for (int nt = 0; nt < 2; ++nt)
            acc[mt][nt] = (f32x8)0.f;

    const int rlo   = lane & 15;
    const int khalf = (lane & 16) ? 8 : 0;

    for (int k0 = 0; k0 < D; k0 += BK) {
        __syncthreads();
        #pragma unroll
        for (int it = 0; it < 8; ++it) {
            int s = it * 256 + tid, row = s >> 4, c4 = s & 15;
            const float4 v = *(const float4*)(X + (size_t)(mBase + row) * D + k0 + c4 * 4);
            uint2 p; p.x = pack_bf16x2(v.x, v.y); p.y = pack_bf16x2(v.z, v.w);
            *(uint2*)&As[row * LDW + c4 * 2] = p;
        }
        #pragma unroll
        for (int it = 0; it < 8; ++it) {
            int s = it * 256 + tid, row = s >> 4, c4 = s & 15;
            const float4 v = *(const float4*)(Xt + (size_t)(nBase + row) * D + k0 + c4 * 4);
            uint2 p; p.x = pack_bf16x2(v.x, v.y); p.y = pack_bf16x2(v.z, v.w);
            *(uint2*)&Bs[row * LDW + c4 * 2] = p;
        }
        __syncthreads();
        #pragma unroll
        for (int kk = 0; kk < BK; kk += 32)
            wmma_step(As, Bs, wrow, wcol, rlo, (kk + khalf) >> 1, acc);
    }

    dist_max_epilogue(acc, w, xsq, ysq, colmax, cmax, mWave, nWave, nBase, lane, wcol, tid);
}

// ---------------------------------------------------------------------------
// Kernel 4: decode encoded column max -> fp32 output (M,1)
// ---------------------------------------------------------------------------
__global__ void decode_out(const unsigned* __restrict__ colmax, float* __restrict__ out, int M) {
    int i = blockIdx.x * 256 + threadIdx.x;
    if (i < M) out[i] = fdec(colmax[i]);
}

// ---------------------------------------------------------------------------
extern "C" void kernel_launch(void* const* d_in, const int* in_sizes, int n_in,
                              void* d_out, int out_size, void* d_ws, size_t ws_size,
                              hipStream_t stream) {
    const float* Xt = (const float*)d_in[0];   // (M, D)
    const float* X  = (const float*)d_in[1];   // (N, D)
    const float* w  = (const float*)d_in[2];   // (N, 1)
    float* out = (float*)d_out;                // (M, 1)

    const int N = in_sizes[2];                 // 32768
    const int M = out_size;                    // 4096
    const int D = in_sizes[1] / N;             // 3072

    float*    xsq    = (float*)d_ws;           // N floats
    float*    ysq    = xsq + N;                // M floats
    unsigned* colmax = (unsigned*)(ysq + M);   // M uints
    unsigned* Xb     = colmax + M;             // N*D/2 uints (bf16 pairs), 16B aligned
    unsigned* Xtb    = Xb + ((size_t)N * D) / 2;

    const size_t needBase = ((size_t)N + 2 * (size_t)M) * sizeof(float);
    const size_t needBf16 = ((size_t)N + (size_t)M) * (size_t)D * 2;
    const bool   fastPath = ws_size >= needBase + needBf16;   // constant across calls

    init_colmax<<<(M + 255) / 256, 256, 0, stream>>>(colmax, M);
    row_sumsq<<<N + M, 256, 0, stream>>>(X, Xt, xsq, ysq, N, M, D);

    dim3 grid(M / BN, N / BM);                 // (32, 256): M fast -> X slab reused in L2
    if (fastPath) {
        const long long nX  = (long long)N * D;
        const long long nXt = (long long)M * D;
        convert_bf16<<<(unsigned)((nX  / 8 + 255) / 256), 256, 0, stream>>>(X,  Xb,  nX);
        convert_bf16<<<(unsigned)((nXt / 8 + 255) / 256), 256, 0, stream>>>(Xt, Xtb, nXt);
        gemm_dist_max_bf16<<<grid, 256, 0, stream>>>(Xb, Xtb, w, xsq, ysq, colmax, D);
    } else {
        gemm_dist_max_f32<<<grid, 256, 0, stream>>>(X, Xt, w, xsq, ysq, colmax, D);
    }

    decode_out<<<(M + 255) / 256, 256, 0, stream>>>(colmax, out, M);
}
